// MultiHeadedAttention_68247030333521
// MI455X (gfx1250) — compile-verified, matrix operand order checked by reading, not run on hardware
//
#include <hip/hip_runtime.h>

// ---------------------------------------------------------------------------
// TENER relative-position multi-head attention for MI455X (gfx1250, wave32)
// bf16 WMMA (v_wmma_f32_16x16x32_bf16) + Tensor Data Mover (tensor_load_to_lds)
// flash-attention fusion, double-buffered TDM GEMM.
// ---------------------------------------------------------------------------

#define DEVI __device__ __forceinline__

typedef __bf16 bf16_t;
typedef __attribute__((ext_vector_type(16))) __bf16 v16bf;
typedef __attribute__((ext_vector_type(8)))  __bf16 v8bf;
typedef __attribute__((ext_vector_type(8)))  float  v8f;
typedef __attribute__((ext_vector_type(4)))  unsigned int u32x4;
typedef __attribute__((ext_vector_type(8)))  int i32x8;
typedef __attribute__((ext_vector_type(4)))  int i32x4;

#define WMMA_BF16(a, b, c) \
  __builtin_amdgcn_wmma_f32_16x16x32_bf16(false, (a), false, (b), (short)0, (c), false, false)

static constexpr int B_  = 2;
static constexpr int S_  = 1024;
static constexpr int D_  = 1024;
static constexpr int H_  = 16;
static constexpr int HD  = 64;      // head dim (also value head dim)
static constexpr int POS_ROWS = 2 * S_;     // 2048 real rows
static constexpr int POS_PAD  = 2064;       // padded (band reads up to row 2048)

// ---------------------------------------------------------------------------
// Tensor Data Mover: 2D tile (tile_d0 elems/row, tile_d1 rows, bf16) from
// global (row stride = stride_elems) into LDS (contiguous row-major tile).
// D# layout per CDNA5 ISA ch.8: group0 = {count, lds_addr, global_addr, type},
// group1 = {data_size, tensor dims, tile dims, dim0 stride}. Groups 2/3 zero
// (2D tile). Issue from ONE wave only (TDM ignores EXEC, issues per-wave).
// ---------------------------------------------------------------------------
DEVI void tdm_load_2d(unsigned lds_off, const void* gaddr,
                      unsigned tile_d0, unsigned tile_d1, unsigned stride_elems) {
  unsigned long long ga = (unsigned long long)(size_t)gaddr;
  u32x4 g0;
  g0[0] = 1u;                                               // count=1, user desc
  g0[1] = lds_off;                                          // lds_addr (bytes)
  g0[2] = (unsigned)(ga & 0xffffffffu);                     // global_addr lo
  g0[3] = (unsigned)((ga >> 32) & 0x1ffffffu) | (2u << 30); // addr hi | type=2
  i32x8 g1;
  g1[0] = (int)(1u << 16);            // workgroup_mask=0, data_size=1 (2 bytes)
  g1[1] = (int)(tile_d0 << 16);       // tensor_dim0[15:0]  (== tile width)
  g1[2] = (int)(tile_d1 << 16);       // tensor_dim1[15:0]  (== tile rows)
  g1[3] = (int)(tile_d0 << 16);       // tile_dim0
  g1[4] = (int)tile_d1;               // tile_dim1 (tile_dim2 = 0)
  g1[5] = (int)stride_elems;          // tensor_dim0_stride lo (data_size units)
  g1[6] = 0;                          // stride hi / dim1 stride (unused, 2D)
  g1[7] = 0;
  i32x4 z4 = {0, 0, 0, 0};
#if __clang_major__ >= 23
  i32x8 z8 = {0, 0, 0, 0, 0, 0, 0, 0};
  __builtin_amdgcn_tensor_load_to_lds(g0, g1, z4, z4, z8, 0);
#else
  __builtin_amdgcn_tensor_load_to_lds(g0, g1, z4, z4, 0);
#endif
}

DEVI unsigned lds_offset(const void* p) {   // low 32 bits of LDS generic addr
  return (unsigned)(size_t)p;
}

// ---------------------------------------------------------------------------
// fp32 -> bf16 cast
// ---------------------------------------------------------------------------
__global__ void cast_f32_to_bf16(const float* __restrict__ src,
                                 bf16_t* __restrict__ dst, int n) {
  int i = blockIdx.x * blockDim.x + threadIdx.x;
  if (i < n) dst[i] = (bf16_t)src[i];
}

// ---------------------------------------------------------------------------
// TENER positional embeddings, bf16, padded with zero rows
// ---------------------------------------------------------------------------
__global__ void pos_embed_kernel(bf16_t* __restrict__ pos) {
  int i = blockIdx.x * blockDim.x + threadIdx.x;
  if (i >= POS_PAD * HD) return;
  int l = i / HD, j = i % HD;
  float val = 0.f;
  if (l < POS_ROWS) {
    int t = (j < 32) ? j : j - 32;
    float div = __expf((float)t * (-9.210340371976184f / 31.0f)); // -ln(1e4)/31
    float ang = (float)(l - S_) * div;
    val = (j < 32) ? __sinf(ang) : __cosf(ang);
  }
  pos[i] = (bf16_t)val;
}

// ---------------------------------------------------------------------------
// Fragment loader: 16 bf16 per lane = two aligned 16B chunks
// ---------------------------------------------------------------------------
DEVI v16bf load_frag(const bf16_t* p0, const bf16_t* p1) {
  v8bf lo = *(const v8bf*)p0;
  v8bf hi = *(const v8bf*)p1;
  v16bf r;
#pragma unroll
  for (int i = 0; i < 8; ++i) { r[i] = lo[i]; r[i + 8] = hi[i]; }
  return r;
}

// ---------------------------------------------------------------------------
// GEMM: out[M,N] = A[M,K](bf16) @ W[N,K]^T(bf16) + bias
// Block tile 128(M) x 128(N), K-step 32, TDM double-buffered LDS staging.
// 256 threads = 8 waves; wave w owns rows [16w,16w+16) -> 8 WMMAs / K-step.
// ---------------------------------------------------------------------------
__global__ __launch_bounds__(256)
void gemm_bf16_wmma(const bf16_t* __restrict__ A, const bf16_t* __restrict__ W,
                    const float* __restrict__ bias, const float* __restrict__ vbias,
                    bf16_t* __restrict__ outb, bf16_t* __restrict__ outb2,
                    float* __restrict__ outf, int M, int N, int K) {
  __shared__ bf16_t As[2][128 * 32];
  __shared__ bf16_t Bs[2][128 * 32];

  const int tid = threadIdx.x, w = tid >> 5, lane = tid & 31;
  const int m15 = lane & 15, g = lane >> 4;
  const int bm = blockIdx.y * 128, bn = blockIdx.x * 128;

  v8f acc[8];
#pragma unroll
  for (int i = 0; i < 8; ++i) { v8f z = {}; acc[i] = z; }

  const int nk = K / 32;
  if (tid < 32) {   // wave 0 drives the Tensor Data Mover
    tdm_load_2d(lds_offset(&As[0][0]), &A[(size_t)bm * K], 32, 128, (unsigned)K);
    tdm_load_2d(lds_offset(&Bs[0][0]), &W[(size_t)bn * K], 32, 128, (unsigned)K);
    __builtin_amdgcn_s_wait_tensorcnt(0);
  }
  __syncthreads();

  for (int i = 0; i < nk; ++i) {
    const int cur = i & 1, nxt = cur ^ 1;
    if (tid < 32 && i + 1 < nk) {   // prefetch next K tile while computing
      tdm_load_2d(lds_offset(&As[nxt][0]), &A[(size_t)bm * K + (i + 1) * 32],
                  32, 128, (unsigned)K);
      tdm_load_2d(lds_offset(&Bs[nxt][0]), &W[(size_t)bn * K + (i + 1) * 32],
                  32, 128, (unsigned)K);
    }

    v16bf afrag = load_frag(&As[cur][(16 * w + m15) * 32 + 8 * g],
                            &As[cur][(16 * w + m15) * 32 + 16 + 8 * g]);
#pragma unroll
    for (int nt = 0; nt < 8; ++nt) {
      v16bf bfrag = load_frag(&Bs[cur][(16 * nt + m15) * 32 + 8 * g],
                              &Bs[cur][(16 * nt + m15) * 32 + 16 + 8 * g]);
      acc[nt] = WMMA_BF16(afrag, bfrag, acc[nt]);
    }

    if (tid < 32) __builtin_amdgcn_s_wait_tensorcnt(0);
    __syncthreads();
  }

#pragma unroll
  for (int nt = 0; nt < 8; ++nt)
#pragma unroll
    for (int r = 0; r < 8; ++r) {
      int row = bm + 16 * w + r + 8 * g;
      int col = bn + 16 * nt + m15;
      float val = acc[nt][r] + bias[col];
      size_t idx = (size_t)row * N + col;
      if (outb)  outb[idx]  = (bf16_t)val;
      if (outb2) outb2[idx] = (bf16_t)(val + vbias[col]);
      if (outf)  outf[idx]  = val;
    }
}

// ---------------------------------------------------------------------------
// Flash attention with TENER relative positions.
//   score[q,k] = q.k  +  (q + v_bias) . pos[S + k - q]
// One block = (b,h, 128 query rows); wave w owns 16 query rows. Keys are
// processed in chunks of 32: K tile staged via TDM, V transposed manually.
// ---------------------------------------------------------------------------
__global__ __launch_bounds__(256)
void tener_attention(const bf16_t* __restrict__ Q, const bf16_t* __restrict__ Qb,
                     const bf16_t* __restrict__ Kmat, const bf16_t* __restrict__ V,
                     const bf16_t* __restrict__ Pos, bf16_t* __restrict__ X) {
  __shared__ bf16_t Kt[32 * 64];        // 32 keys x 64 dims (TDM destination)
  __shared__ bf16_t Vt[64 * 32];        // transposed: 64 dims x 32 keys
  __shared__ float  Rb[8][16 * 48];     // per-wave rel band
  __shared__ bf16_t Pb[8][16 * 32];     // per-wave probs (A-layout staging)

  const int tid = threadIdx.x, w = tid >> 5, lane = tid & 31;
  const int m15 = lane & 15, g = lane >> 4;
  const int qblkN = S_ / 128;                       // 8
  const int qblk = blockIdx.x % qblkN;
  const int h = (blockIdx.x / qblkN) % H_;
  const int b = blockIdx.x / (qblkN * H_);
  const int q0 = qblk * 128 + 16 * w;

  const size_t qrow = ((size_t)(b * S_ + q0 + m15) * H_ + h) * HD;
  v16bf qf[2], qbf2[2];
#pragma unroll
  for (int ks = 0; ks < 2; ++ks) {
    qf[ks]   = load_frag(&Q[qrow + 32 * ks + 8 * g],  &Q[qrow + 32 * ks + 16 + 8 * g]);
    qbf2[ks] = load_frag(&Qb[qrow + 32 * ks + 8 * g], &Qb[qrow + 32 * ks + 16 + 8 * g]);
  }

  float mrow[8], lrow[8];
  v8f oacc[4];
#pragma unroll
  for (int r = 0; r < 8; ++r) { mrow[r] = -3.0e38f; lrow[r] = 0.f; }
#pragma unroll
  for (int i = 0; i < 4; ++i) { v8f z = {}; oacc[i] = z; }

  for (int c = 0; c < S_ / 32; ++c) {
    const int k0 = 32 * c;
    // K tile via Tensor Data Mover (wave 0), V tile transposed by all threads
    if (tid < 32)
      tdm_load_2d(lds_offset(&Kt[0]),
                  &Kmat[((size_t)(b * S_ + k0) * H_ + h) * HD],
                  64, 32, (unsigned)(H_ * HD));
    {
      int kr = tid >> 3, ch = tid & 7;
      v8bf vv = *(const v8bf*)&V[((size_t)(b * S_ + k0 + kr) * H_ + h) * HD + ch * 8];
#pragma unroll
      for (int e = 0; e < 8; ++e) Vt[(ch * 8 + e) * 32 + kr] = vv[e];
    }
    if (tid < 32) __builtin_amdgcn_s_wait_tensorcnt(0);
    __syncthreads();

    // ---- scores: Q(16x64) . K^T -> two 16x16 tiles
    v8f s[2]; { v8f z = {}; s[0] = z; s[1] = z; }
#pragma unroll
    for (int t = 0; t < 2; ++t)
#pragma unroll
      for (int ks = 0; ks < 2; ++ks) {
        v16bf kf = load_frag(&Kt[(16 * t + m15) * 64 + 32 * ks + 8 * g],
                             &Kt[(16 * t + m15) * 64 + 32 * ks + 16 + 8 * g]);
        s[t] = WMMA_BF16(qf[ks], kf, s[t]);
      }

    // ---- relative band: R[m][cc] = (q+v_bias) . pos[lbase+cc], cc in [0,48)
    const int lbase = S_ + k0 - q0 - 15;   // in [1, 2S-47]; pad covers +47
#pragma unroll
    for (int t = 0; t < 3; ++t) {
      v8f ra; { v8f z = {}; ra = z; }
#pragma unroll
      for (int ks = 0; ks < 2; ++ks) {
        const size_t prow = (size_t)(lbase + 16 * t + m15) * HD;
        v16bf pf = load_frag(&Pos[prow + 32 * ks + 8 * g],
                             &Pos[prow + 32 * ks + 16 + 8 * g]);
        ra = WMMA_BF16(qbf2[ks], pf, ra);
      }
#pragma unroll
      for (int r = 0; r < 8; ++r)
        Rb[w][(r + 8 * g) * 48 + 16 * t + m15] = ra[r];
    }
    // gather shift: rel[m][j] = R[m][j - m + 15]
#pragma unroll
    for (int t = 0; t < 2; ++t)
#pragma unroll
      for (int r = 0; r < 8; ++r) {
        int mi = r + 8 * g;
        int j  = 16 * t + m15;
        s[t][r] += Rb[w][mi * 48 + (j - mi + 15)];
      }

    // ---- online softmax (row m = r+8g lives in 16 lanes of half g)
#pragma unroll
    for (int r = 0; r < 8; ++r) {
      float mx = fmaxf(s[0][r], s[1][r]);
#pragma unroll
      for (int d = 1; d < 16; d <<= 1) mx = fmaxf(mx, __shfl_xor(mx, d, 32));
      float mnew  = fmaxf(mrow[r], mx);
      float scale = __expf(mrow[r] - mnew);
      float p0 = __expf(s[0][r] - mnew);
      float p1 = __expf(s[1][r] - mnew);
      s[0][r] = p0; s[1][r] = p1;
      float rs = p0 + p1;
#pragma unroll
      for (int d = 1; d < 16; d <<= 1) rs += __shfl_xor(rs, d, 32);
      lrow[r] = lrow[r] * scale + rs;
      mrow[r] = mnew;
#pragma unroll
      for (int nt = 0; nt < 4; ++nt) oacc[nt][r] *= scale;
    }

    // ---- restage P (D-layout -> A-layout) through LDS, then P @ V
#pragma unroll
    for (int t = 0; t < 2; ++t)
#pragma unroll
      for (int r = 0; r < 8; ++r)
        Pb[w][(r + 8 * g) * 32 + 16 * t + m15] = (bf16_t)s[t][r];
    v16bf pfrag = load_frag(&Pb[w][m15 * 32 + 8 * g],
                            &Pb[w][m15 * 32 + 16 + 8 * g]);
#pragma unroll
    for (int nt = 0; nt < 4; ++nt) {
      v16bf vf = load_frag(&Vt[(16 * nt + m15) * 32 + 8 * g],
                           &Vt[(16 * nt + m15) * 32 + 16 + 8 * g]);
      oacc[nt] = WMMA_BF16(pfrag, vf, oacc[nt]);
    }
    __syncthreads();
  }

  // normalize + store x in [B,S,H,dv] (== reference [B,S,Dv] layout)
#pragma unroll
  for (int nt = 0; nt < 4; ++nt)
#pragma unroll
    for (int r = 0; r < 8; ++r) {
      int srow = q0 + r + 8 * g;
      size_t idx = ((size_t)(b * S_ + srow) * H_ + h) * HD + 16 * nt + m15;
      X[idx] = (bf16_t)(oacc[nt][r] / lrow[r]);
    }
}

// ---------------------------------------------------------------------------
extern "C" void kernel_launch(void* const* d_in, const int* in_sizes, int n_in,
                              void* d_out, int out_size, void* d_ws, size_t ws_size,
                              hipStream_t stream) {
  (void)in_sizes; (void)n_in; (void)out_size; (void)ws_size;

  const float* query = (const float*)d_in[0];
  const float* key   = (const float*)d_in[1];
  const float* value = (const float*)d_in[2];
  /* d_in[3] = mask (all ones; reference softmax is unmasked) */
  const float* Wq    = (const float*)d_in[4];
  const float* bq    = (const float*)d_in[5];
  const float* Wv    = (const float*)d_in[6];
  const float* bv    = (const float*)d_in[7];
  const float* Wo    = (const float*)d_in[8];
  const float* bo    = (const float*)d_in[9];
  const float* vbias = (const float*)d_in[10];
  float* out = (float*)d_out;

  char* ws = (char*)d_ws;
  const size_t MB = (size_t)1 << 20;
  bf16_t* query_bf = (bf16_t*)(ws + 0 * MB);
  bf16_t* key_bf   = (bf16_t*)(ws + 4 * MB);   // == k in [B,S,H,d] layout
  bf16_t* value_bf = (bf16_t*)(ws + 8 * MB);
  bf16_t* Wq_bf    = (bf16_t*)(ws + 12 * MB);
  bf16_t* Wv_bf    = (bf16_t*)(ws + 14 * MB);
  bf16_t* Wo_bf    = (bf16_t*)(ws + 16 * MB);
  bf16_t* q_bf     = (bf16_t*)(ws + 18 * MB);
  bf16_t* qb_bf    = (bf16_t*)(ws + 22 * MB);  // q + v_bias
  bf16_t* v_bf     = (bf16_t*)(ws + 26 * MB);
  bf16_t* x_bf     = (bf16_t*)(ws + 30 * MB);
  bf16_t* pos_bf   = (bf16_t*)(ws + 34 * MB);

  const int n2 = B_ * S_ * D_;   // 2M
  const int n1 = D_ * D_;        // 1M
  cast_f32_to_bf16<<<n2 / 256, 256, 0, stream>>>(query, query_bf, n2);
  cast_f32_to_bf16<<<n2 / 256, 256, 0, stream>>>(key,   key_bf,   n2);
  cast_f32_to_bf16<<<n2 / 256, 256, 0, stream>>>(value, value_bf, n2);
  cast_f32_to_bf16<<<n1 / 256, 256, 0, stream>>>(Wq, Wq_bf, n1);
  cast_f32_to_bf16<<<n1 / 256, 256, 0, stream>>>(Wv, Wv_bf, n1);
  cast_f32_to_bf16<<<n1 / 256, 256, 0, stream>>>(Wo, Wo_bf, n1);
  pos_embed_kernel<<<(POS_PAD * HD) / 256, 256, 0, stream>>>(pos_bf);

  dim3 ggrid(D_ / 128, (B_ * S_) / 128);
  // q = query@Wq^T + bq (and qb = q + v_bias), both bf16
  gemm_bf16_wmma<<<ggrid, 256, 0, stream>>>(query_bf, Wq_bf, bq, vbias,
                                            q_bf, qb_bf, nullptr,
                                            B_ * S_, D_, D_);
  // v = value@Wv^T + bv
  gemm_bf16_wmma<<<ggrid, 256, 0, stream>>>(value_bf, Wv_bf, bv, nullptr,
                                            v_bf, nullptr, nullptr,
                                            B_ * S_, D_, D_);
  // fused flash attention with relative positions
  tener_attention<<<B_ * H_ * (S_ / 128), 256, 0, stream>>>(
      q_bf, qb_bf, key_bf, v_bf, pos_bf, x_bf);
  // out = x@Wo^T + bo, fp32 straight to d_out
  gemm_bf16_wmma<<<ggrid, 256, 0, stream>>>(x_bf, Wo_bf, bo, nullptr,
                                            nullptr, nullptr, out,
                                            B_ * S_, D_, D_);
}
